// R2D2Network_69432441307616
// MI455X (gfx1250) — matrix operand order repaired and checked by reading, not compiled, not run on hardware
//
#include <hip/hip_runtime.h>
#include <math.h>

#define TT 128
#define BB 256
#define OBSZ 128
#define HH 512
#define AA 18
#define NROW (TT * BB)     // 32768 sample rows
#define G4H (4 * HH)       // 2048 gate units
#define LSTMIN (HH + AA + 1)

// ---------------- CDNA5 WMMA types ----------------
typedef __bf16 v16bf __attribute__((ext_vector_type(16)));
typedef float  v8f   __attribute__((ext_vector_type(8)));
typedef unsigned int u32x4 __attribute__((ext_vector_type(4)));

union ABFrag { v16bf v; u32x4 q[2]; };
union CDFrag { v8f v; float f[8]; };

__device__ __forceinline__ unsigned short f2bf(float x) {
  unsigned int u = __float_as_uint(x);
  u += 0x7FFFu + ((u >> 16) & 1u);       // round-to-nearest-even
  return (unsigned short)(u >> 16);
}

__device__ __forceinline__ float sigmoidf_(float x) {
  return 1.0f / (1.0f + expf(-x));
}

// Wave computes one 16(M) x 64(N) tile, K-loop in steps of 32 (bf16 WMMA).
// A:  bf16 [M,K] row-major.  Bt: bf16 [N,K] row-major (weights pre-transposed).
// CDNA5 16-bit A frag per lane (m=lane&15, h=lane>>4): K in [kb+8h, kb+8h+8) U [kb+16+8h, +8)
// B frag per lane (n=lane&15, h=lane>>4): K in [kb+8h, kb+8h+8) U [kb+16+8h, +8) (same as A).
__device__ __forceinline__ void wmma_block_16x64(
    const unsigned short* __restrict__ A, int lda,
    const unsigned short* __restrict__ Bt, int ldb,
    int K, int mBase, int nBase, int lane, CDFrag acc[4])
{
  const int l16 = lane & 15;
  const int h   = lane >> 4;
  const unsigned short* aRow = A  + (size_t)(mBase + l16) * lda;
  const unsigned short* bRow = Bt + (size_t)(nBase + l16) * ldb + 8 * h;
  for (int kb = 0; kb < K; kb += 32) {
    ABFrag a;
    a.q[0] = *(const u32x4*)(aRow + kb + 8 * h);
    a.q[1] = *(const u32x4*)(aRow + kb + 16 + 8 * h);
#pragma unroll
    for (int t = 0; t < 4; ++t) {
      ABFrag b;
      const unsigned short* bp = bRow + (size_t)t * 16 * ldb + kb;
      b.q[0] = *(const u32x4*)(bp);
      b.q[1] = *(const u32x4*)(bp + 16);
      acc[t].v = __builtin_amdgcn_wmma_f32_16x16x32_bf16(
          false, a.v, false, b.v, (short)0, acc[t].v, false, false);
    }
  }
}

// Generic GEMM: grid = (N/256, M/16), block = 128 (4 waves).
// out = [relu]( A @ Bt^T + bias + Cinit ); writes f32 and/or bf16.
__global__ void __launch_bounds__(128)
wmma_gemm(const unsigned short* __restrict__ A, int lda,
          const unsigned short* __restrict__ Bt, int ldb,
          const float* __restrict__ Cinit, int ldci,
          const float* __restrict__ bias,
          float* __restrict__ outF, unsigned short* __restrict__ outB, int ldc,
          int K, int relu)
{
  const int lane  = threadIdx.x & 31;
  const int wave  = threadIdx.x >> 5;
  const int mBase = blockIdx.y * 16;
  const int nBase = blockIdx.x * 256 + wave * 64;
  CDFrag acc[4];
  const v8f zero = {0.f, 0.f, 0.f, 0.f, 0.f, 0.f, 0.f, 0.f};
#pragma unroll
  for (int t = 0; t < 4; ++t) acc[t].v = zero;

  wmma_block_16x64(A, lda, Bt, ldb, K, mBase, nBase, lane, acc);

  const int l16 = lane & 15;
  const int h   = lane >> 4;
#pragma unroll
  for (int t = 0; t < 4; ++t) {
    const int col = nBase + t * 16 + l16;
    const float bi = bias ? bias[col] : 0.0f;
#pragma unroll
    for (int j = 0; j < 8; ++j) {
      const int row = mBase + j + 8 * h;
      float v = acc[t].f[j] + bi;
      if (Cinit) v += Cinit[(size_t)row * ldci + col];
      if (relu) v = v > 0.0f ? v : 0.0f;
      const size_t oi = (size_t)row * ldc + col;
      if (outF) outF[oi] = v;
      if (outB) outB[oi] = f2bf(v);
    }
  }
}

// xg = hidden @ W_ih[:, :512]^T  +  r*W_ih[:,512] + W_ih[:,513+a] + b_ih + b_hh
// grid = (2048/256, 32768/16), block 128.
__global__ void __launch_bounds__(128)
wmma_gemm_xg(const unsigned short* __restrict__ A,     // [32768,512] bf16
             const unsigned short* __restrict__ Bt,    // [2048,512] bf16
             const float* __restrict__ W_ih,           // [2048,531] f32
             const float* __restrict__ b_ih, const float* __restrict__ b_hh,
             const float* __restrict__ r, const int* __restrict__ act,
             float* __restrict__ xg)
{
  const int lane  = threadIdx.x & 31;
  const int wave  = threadIdx.x >> 5;
  const int mBase = blockIdx.y * 16;
  const int nBase = blockIdx.x * 256 + wave * 64;
  CDFrag acc[4];
  const v8f zero = {0.f, 0.f, 0.f, 0.f, 0.f, 0.f, 0.f, 0.f};
#pragma unroll
  for (int t = 0; t < 4; ++t) acc[t].v = zero;

  wmma_block_16x64(A, HH, Bt, HH, HH, mBase, nBase, lane, acc);

  const int l16 = lane & 15;
  const int h   = lane >> 4;
#pragma unroll
  for (int t = 0; t < 4; ++t) {
    const int g = nBase + t * 16 + l16;               // gate index
    const float* wrow = W_ih + (size_t)g * LSTMIN;
    const float wg   = wrow[HH];                      // column for r
    const float bsum = b_ih[g] + b_hh[g];
#pragma unroll
    for (int j = 0; j < 8; ++j) {
      const int row = mBase + j + 8 * h;              // sample index
      float v = acc[t].f[j] + r[row] * wg + wrow[HH + 1 + act[row]] + bsum;
      xg[(size_t)row * G4H + g] = v;
    }
  }
}

// ---------------- conversion / elementwise kernels ----------------
__global__ void conv_bf(const float* __restrict__ in, unsigned short* __restrict__ out,
                        int rows, int cols, int stride)
{
  int idx = blockIdx.x * blockDim.x + threadIdx.x;
  if (idx >= rows * cols) return;
  int rr = idx / cols, cc = idx - rr * cols;
  out[idx] = f2bf(in[(size_t)rr * stride + cc]);
}

__global__ void transpose_bf(const float* __restrict__ in, unsigned short* __restrict__ out,
                             int R, int C)   // in [R,C] -> out [C,R]
{
  int idx = blockIdx.x * blockDim.x + threadIdx.x;
  if (idx >= R * C) return;
  int rr = idx / C, cc = idx - rr * C;
  out[(size_t)cc * R + rr] = f2bf(in[idx]);
}

__global__ void init_state(const float* __restrict__ h0, const float* __restrict__ c0,
                           const float* __restrict__ done0,
                           unsigned short* __restrict__ h_bf, float* __restrict__ c)
{
  int idx = blockIdx.x * blockDim.x + threadIdx.x;
  if (idx >= BB * HH) return;
  int b = idx >> 9;
  float keep = 1.0f - done0[b];
  h_bf[idx] = f2bf(h0[idx] * keep);
  c[idx] = c0[idx] * keep;
}

__global__ void lstm_update(const float* __restrict__ gates,   // [256,2048]
                            float* __restrict__ c,             // [256,512] in/out (masked)
                            const float* __restrict__ done_next, // [256] or null
                            unsigned short* __restrict__ h_bf, // next-step A matrix
                            unsigned short* __restrict__ hs_bf_t, // hs slice (unmasked)
                            float* __restrict__ hT, float* __restrict__ cT) // last step only
{
  int idx = blockIdx.x * blockDim.x + threadIdx.x;
  if (idx >= BB * HH) return;
  int b = idx >> 9;
  int j = idx & (HH - 1);
  const float* g = gates + (size_t)b * G4H;
  float gi = g[j], gf = g[HH + j], gg = g[2 * HH + j], go = g[3 * HH + j];
  float cNew = sigmoidf_(gf) * c[idx] + sigmoidf_(gi) * tanhf(gg);
  float hNew = sigmoidf_(go) * tanhf(cNew);
  hs_bf_t[idx] = f2bf(hNew);
  if (hT) { hT[idx] = hNew; cT[idx] = cNew; }
  float keep = done_next ? (1.0f - done_next[b]) : 1.0f;
  h_bf[idx] = f2bf(hNew * keep);
  c[idx]    = cNew * keep;
}

// adv[n,a] = ab1[a] + ah[n,:] . aW1[:,a]   (one wave per row, lanes 0..17)
__global__ void adv_kernel(const float* __restrict__ ah, const float* __restrict__ aW1,
                           const float* __restrict__ ab1, float* __restrict__ adv)
{
  int wid  = (blockIdx.x * blockDim.x + threadIdx.x) >> 5;
  int lane = threadIdx.x & 31;
  if (wid >= NROW || lane >= AA) return;
  const float* row = ah + (size_t)wid * HH;
  float acc = ab1[lane];
  for (int k = 0; k < HH; ++k) acc += row[k] * aW1[k * AA + lane];
  adv[(size_t)wid * AA + lane] = acc;
}

// val[n] = vb1 + vh[n,:] . vW1   (one wave per row, lane-strided + shuffle reduce)
__global__ void val_kernel(const float* __restrict__ vh, const float* __restrict__ vW1,
                           const float* __restrict__ vb1, float* __restrict__ val)
{
  int wid  = (blockIdx.x * blockDim.x + threadIdx.x) >> 5;
  int lane = threadIdx.x & 31;
  if (wid >= NROW) return;
  const float* row = vh + (size_t)wid * HH;
  float acc = 0.0f;
  for (int k = lane; k < HH; k += 32) acc += row[k] * vW1[k];
  for (int o = 16; o > 0; o >>= 1) acc += __shfl_down(acc, o, 32);
  if (lane == 0) val[wid] = acc + vb1[0];
}

__global__ void q_kernel(const float* __restrict__ adv, const float* __restrict__ val,
                         float* __restrict__ q)
{
  int n = blockIdx.x * blockDim.x + threadIdx.x;
  if (n >= NROW) return;
  const float* ar = adv + (size_t)n * AA;
  float m = 0.0f;
#pragma unroll
  for (int a = 0; a < AA; ++a) m += ar[a];
  m *= (1.0f / AA);
  float v = val[n];
  float* qr = q + (size_t)n * AA;
#pragma unroll
  for (int a = 0; a < AA; ++a) qr[a] = v + ar[a] - m;
}

// ---------------- host orchestration ----------------
extern "C" void kernel_launch(void* const* d_in, const int* in_sizes, int n_in,
                              void* d_out, int out_size, void* d_ws, size_t ws_size,
                              hipStream_t stream)
{
  (void)in_sizes; (void)n_in; (void)out_size; (void)ws_size;
  const float* o    = (const float*)d_in[0];
  const int*   a    = (const int*)  d_in[1];
  const float* r    = (const float*)d_in[2];
  const float* done = (const float*)d_in[3];
  const float* h0   = (const float*)d_in[4];
  const float* c0   = (const float*)d_in[5];
  const float* tW0  = (const float*)d_in[6];
  const float* tb0  = (const float*)d_in[7];
  const float* tW1  = (const float*)d_in[8];
  const float* tb1  = (const float*)d_in[9];
  const float* W_ih = (const float*)d_in[10];
  const float* W_hh = (const float*)d_in[11];
  const float* b_ih = (const float*)d_in[12];
  const float* b_hh = (const float*)d_in[13];
  const float* aW0  = (const float*)d_in[14];
  const float* ab0  = (const float*)d_in[15];
  const float* aW1  = (const float*)d_in[16];
  const float* ab1  = (const float*)d_in[17];
  const float* vW0  = (const float*)d_in[18];
  const float* vb0  = (const float*)d_in[19];
  const float* vW1  = (const float*)d_in[20];
  const float* vb1  = (const float*)d_in[21];

  float* qOut = (float*)d_out;
  float* hT   = qOut + (size_t)NROW * AA;
  float* cT   = hT + (size_t)BB * HH;

  char* ws = (char*)d_ws;
  size_t off = 0;
  auto alloc = [&](size_t bytes) -> void* {
    void* p = ws + off;
    off += (bytes + 255) & ~(size_t)255;
    return p;
  };
  unsigned short* o_bf   = (unsigned short*)alloc((size_t)NROW * OBSZ * 2);
  unsigned short* tW0t   = (unsigned short*)alloc((size_t)HH * OBSZ * 2);   // [512,128]
  unsigned short* tW1t   = (unsigned short*)alloc((size_t)HH * HH * 2);     // [512,512]
  unsigned short* Wih_bf = (unsigned short*)alloc((size_t)G4H * HH * 2);    // [2048,512]
  unsigned short* Whh_bf = (unsigned short*)alloc((size_t)G4H * HH * 2);    // [2048,512]
  unsigned short* aW0t   = (unsigned short*)alloc((size_t)HH * HH * 2);
  unsigned short* vW0t   = (unsigned short*)alloc((size_t)HH * HH * 2);
  unsigned short* hid0   = (unsigned short*)alloc((size_t)NROW * HH * 2);   // reused as hs_bf
  unsigned short* hid1   = (unsigned short*)alloc((size_t)NROW * HH * 2);
  float*          xg     = (float*)alloc((size_t)NROW * G4H * 4);           // 268 MB, reused for heads
  unsigned short* h_bf   = (unsigned short*)alloc((size_t)BB * HH * 2);
  float*          c_cur  = (float*)alloc((size_t)BB * HH * 4);
  float*          gates  = (float*)alloc((size_t)BB * G4H * 4);
  float*          advB   = (float*)alloc((size_t)NROW * AA * 4);
  float*          valB   = (float*)alloc((size_t)NROW * 4);

  unsigned short* hs_bf = hid0;            // trunk hidden0 is dead once scan starts
  float* ahF = xg;                         // xg dead after scan
  float* vhF = xg + (size_t)NROW * HH;

  const int thr = 256;
  const dim3 gblk(128);

  // --- weight / input conversions to bf16 (weights in transposed [N,K] layout) ---
  conv_bf<<<(NROW * OBSZ + thr - 1) / thr, thr, 0, stream>>>(o, o_bf, NROW, OBSZ, OBSZ);
  transpose_bf<<<(OBSZ * HH + thr - 1) / thr, thr, 0, stream>>>(tW0, tW0t, OBSZ, HH);
  transpose_bf<<<(HH * HH + thr - 1) / thr, thr, 0, stream>>>(tW1, tW1t, HH, HH);
  conv_bf<<<(G4H * HH + thr - 1) / thr, thr, 0, stream>>>(W_ih, Wih_bf, G4H, HH, LSTMIN);
  conv_bf<<<(G4H * HH + thr - 1) / thr, thr, 0, stream>>>(W_hh, Whh_bf, G4H, HH, HH);
  transpose_bf<<<(HH * HH + thr - 1) / thr, thr, 0, stream>>>(aW0, aW0t, HH, HH);
  transpose_bf<<<(HH * HH + thr - 1) / thr, thr, 0, stream>>>(vW0, vW0t, HH, HH);

  // --- trunk: two Linear+ReLU on WMMA ---
  wmma_gemm<<<dim3(HH / 256, NROW / 16), gblk, 0, stream>>>(
      o_bf, OBSZ, tW0t, OBSZ, nullptr, 0, tb0, nullptr, hid0, HH, OBSZ, 1);
  wmma_gemm<<<dim3(HH / 256, NROW / 16), gblk, 0, stream>>>(
      hid0, HH, tW1t, HH, nullptr, 0, tb1, nullptr, hid1, HH, HH, 1);

  // --- input-gate precompute xg (GEMM + one-hot/r/bias epilogue) ---
  wmma_gemm_xg<<<dim3(G4H / 256, NROW / 16), gblk, 0, stream>>>(
      hid1, Wih_bf, W_ih, b_ih, b_hh, r, a, xg);

  // --- LSTM scan: sequential over T; each step is a WMMA GEMM + gate update ---
  init_state<<<(BB * HH) / thr, thr, 0, stream>>>(h0, c0, done, h_bf, c_cur);
  for (int t = 0; t < TT; ++t) {
    wmma_gemm<<<dim3(G4H / 256, BB / 16), gblk, 0, stream>>>(
        h_bf, HH, Whh_bf, HH, xg + (size_t)t * BB * G4H, G4H,
        nullptr, gates, nullptr, G4H, HH, 0);
    lstm_update<<<(BB * HH) / thr, thr, 0, stream>>>(
        gates, c_cur,
        (t + 1 < TT) ? (done + (size_t)(t + 1) * BB) : nullptr,
        h_bf, hs_bf + (size_t)t * BB * HH,
        (t == TT - 1) ? hT : nullptr, (t == TT - 1) ? cT : nullptr);
  }

  // --- dueling heads ---
  wmma_gemm<<<dim3(HH / 256, NROW / 16), gblk, 0, stream>>>(
      hs_bf, HH, aW0t, HH, nullptr, 0, ab0, ahF, nullptr, HH, HH, 1);
  adv_kernel<<<(NROW * 32 + thr - 1) / thr, thr, 0, stream>>>(ahF, aW1, ab1, advB);
  wmma_gemm<<<dim3(HH / 256, NROW / 16), gblk, 0, stream>>>(
      hs_bf, HH, vW0t, HH, nullptr, 0, vb0, vhF, nullptr, HH, HH, 1);
  val_kernel<<<(NROW * 32 + thr - 1) / thr, thr, 0, stream>>>(vhF, vW1, vb1, valB);
  q_kernel<<<(NROW + thr - 1) / thr, thr, 0, stream>>>(advB, valB, qOut);
}
